// MultiHeadAttention_61933428416958
// MI455X (gfx1250) — matrix operand order repaired and checked by reading, not compiled, not run on hardware
//
#include <hip/hip_runtime.h>
#include <cstdint>

typedef __attribute__((ext_vector_type(16))) _Float16 v16h;
typedef __attribute__((ext_vector_type(8)))  _Float16 v8h;
typedef __attribute__((ext_vector_type(8)))  float    v8f;
typedef __attribute__((ext_vector_type(4)))  float    v4f;

// D = A(16x32 f16) * B(32x16 f16) + C(16x16 f32)
static __device__ __forceinline__ v8f wmma_f16(v16h a, v16h b, v8f c) {
  return __builtin_amdgcn_wmma_f32_16x16x32_f16(
      /*neg_a=*/false, a, /*neg_b=*/false, b,
      /*c_mod=*/(short)0, c, /*reuse_a=*/false, /*reuse_b=*/false);
}

// A fragment (16x32, 16-bit): lane holds row m=l&15; element j -> k =
// (j>=8?16:0) + (l>=16?8:0) + (j&7).  base = &A[0][0], row stride in halfs.
static __device__ __forceinline__ v16h load_frag_a(const _Float16* base, int stride, int lane) {
  const int m = lane & 15, hi = lane >> 4;
  const _Float16* p = base + (size_t)m * stride + hi * 8;
  v8h lo = *(const v8h*)(p);
  v8h hp = *(const v8h*)(p + 16);
  v16h r;
#pragma unroll
  for (int i = 0; i < 8; ++i) { r[i] = lo[i]; r[i + 8] = hp[i]; }
  return r;
}

// B fragment (32x16, 16-bit): lane holds column n=l&15; element j -> k =
// (j>=8?16:0) + (l>=16?8:0) + (j&7).  Staged column-major: base = &B[n=0][k=0] (stride in halfs).
static __device__ __forceinline__ v16h load_frag_b(const _Float16* base, int stride, int lane) {
  const int n = lane & 15, hi = lane >> 4;
  const _Float16* p = base + (size_t)n * stride + hi * 8;
  v8h lo = *(const v8h*)(p);
  v8h hp = *(const v8h*)(p + 16);
  v16h r;
#pragma unroll
  for (int i = 0; i < 8; ++i) { r[i] = lo[i]; r[i + 8] = hp[i]; }
  return r;
}

static __device__ __forceinline__ unsigned pack2f(float a, float b) {
  union { _Float16 h[2]; unsigned u; } t;
  t.h[0] = (_Float16)a; t.h[1] = (_Float16)b;
  return t.u;
}
static __device__ __forceinline__ unsigned pack2h(_Float16 a, _Float16 b) {
  union { _Float16 h[2]; unsigned u; } t;
  t.h[0] = a; t.h[1] = b;
  return t.u;
}
static __device__ __forceinline__ unsigned lds_off(const void* p) {
  // generic pointer into LDS: addr[31:0] is the wave-relative LDS byte offset
  return (unsigned)(uintptr_t)p;
}

// ---------------------------------------------------------------- GEMM ----
// C[M,N] = A[M,K] * W[K,N]; 128x128 block, BK=32, 256 threads (8 waves, 4x2),
// each wave computes 32x64 (2x4 16x16 tiles).  Double-buffered LDS so tile
// i+1 global loads overlap tile i WMMAs (one barrier per k-step).
template <bool A_F16, bool FUSE_RES>
__global__ __launch_bounds__(256) void gemm_kernel(
    const void* __restrict__ Ap, const float* __restrict__ W,
    _Float16* __restrict__ Ch, float* __restrict__ Cf,
    const float* __restrict__ bias, const float* __restrict__ residual,
    int M, int K, int N) {
  constexpr int AS = 40, BS = 40;            // padded strides (banks), 16B-mult
  __shared__ _Float16 Asm[2][128][AS];
  __shared__ _Float16 Bsm[2][128][BS];       // Bsm[.][n][k] (transposed stage)
  const int tid = threadIdx.x, lane = tid & 31, wave = tid >> 5;
  const int wm = wave >> 1, wn = wave & 1;
  const int mo = blockIdx.y * 128, no = blockIdx.x * 128;

  v8f acc[2][4];
#pragma unroll
  for (int i = 0; i < 2; ++i)
#pragma unroll
    for (int j = 0; j < 4; ++j) acc[i][j] = (v8f)0.0f;

  const int ar = tid >> 1, ac0 = (tid & 1) * 16;  // A: 128 rows x 32 k
  const int bkp = tid >> 4, bn0 = (tid & 15) * 8; // W: 16 k-pairs x 8 n each

  auto stage = [&](int it, int buf) {
    const int kk = it * 32;
    if (A_F16) {
      const _Float16* src = (const _Float16*)Ap + (size_t)(mo + ar) * K + kk + ac0;
      *(v8h*)&Asm[buf][ar][ac0]     = *(const v8h*)(src);
      *(v8h*)&Asm[buf][ar][ac0 + 8] = *(const v8h*)(src + 8);
      if (kk + 64 < K) __builtin_prefetch(src + 64, 0, 1);
    } else {
      const float* src = (const float*)Ap + (size_t)(mo + ar) * K + kk + ac0;
      v4f x0 = *(const v4f*)(src), x1 = *(const v4f*)(src + 4);
      v4f x2 = *(const v4f*)(src + 8), x3 = *(const v4f*)(src + 12);
      v8h a0, a1;
#pragma unroll
      for (int j = 0; j < 4; ++j) {
        a0[j] = (_Float16)x0[j]; a0[4 + j] = (_Float16)x1[j];
        a1[j] = (_Float16)x2[j]; a1[4 + j] = (_Float16)x3[j];
      }
      *(v8h*)&Asm[buf][ar][ac0]     = a0;
      *(v8h*)&Asm[buf][ar][ac0 + 8] = a1;
      if (kk + 64 < K) __builtin_prefetch(src + 64, 0, 1);
    }
    {
      const float* r0 = W + (size_t)(kk + 2 * bkp) * N + no + bn0;
      const float* r1 = r0 + N;
      v4f x00 = *(const v4f*)(r0), x01 = *(const v4f*)(r0 + 4);
      v4f x10 = *(const v4f*)(r1), x11 = *(const v4f*)(r1 + 4);
#pragma unroll
      for (int j = 0; j < 4; ++j) {  // pack (k, k+1) for one n -> one b32 store
        *(unsigned*)&Bsm[buf][bn0 + j][2 * bkp]     = pack2f(x00[j], x10[j]);
        *(unsigned*)&Bsm[buf][bn0 + 4 + j][2 * bkp] = pack2f(x01[j], x11[j]);
      }
      if (kk + 64 < K) __builtin_prefetch(r0 + 64 * (size_t)N, 0, 1);
    }
  };

  const int nk = K / 32;
  stage(0, 0);
  for (int it = 0; it < nk; ++it) {
    const int cur = it & 1;
    __syncthreads();                       // buf[cur] ready; prior reads done
    if (it + 1 < nk) stage(it + 1, cur ^ 1);

    v16h af[2], bf[4];
#pragma unroll
    for (int i = 0; i < 2; ++i) af[i] = load_frag_a(&Asm[cur][wm * 32 + i * 16][0], AS, lane);
#pragma unroll
    for (int j = 0; j < 4; ++j) bf[j] = load_frag_b(&Bsm[cur][wn * 64 + j * 16][0], BS, lane);
#pragma unroll
    for (int i = 0; i < 2; ++i)
#pragma unroll
      for (int j = 0; j < 4; ++j) acc[i][j] = wmma_f16(af[i], bf[j], acc[i][j]);
  }

  const int hi = lane >> 4, nl = lane & 15;
#pragma unroll
  for (int i = 0; i < 2; ++i)
#pragma unroll
    for (int j = 0; j < 4; ++j)
#pragma unroll
      for (int r = 0; r < 8; ++r) {
        const int gm = mo + wm * 32 + i * 16 + r + hi * 8;
        const int gn = no + wn * 64 + j * 16 + nl;
        const float vv = acc[i][j][r];
        if (FUSE_RES) {
          Cf[(size_t)gm * N + gn] = vv + bias[gn] + residual[(size_t)gm * N + gn];
        } else {
          Ch[(size_t)gm * N + gn] = (_Float16)vv;
        }
      }
}

// ----------------------------------------------------------- Attention ----
// One WG per (128 q-rows, b, h). 8 waves, 16 rows each. Flash-style online
// softmax over key chunks of 64.  qh/kh/vh are f16 [B,S,H*64].
// K chunk is a pure f16 copy -> CDNA5 async global->LDS path (ASYNCcnt).
__global__ __launch_bounds__(256) void attn_kernel(
    const _Float16* __restrict__ qh, const _Float16* __restrict__ kh,
    const _Float16* __restrict__ vh, _Float16* __restrict__ outh, int S) {
  constexpr int KS = 80;                       // padded stride (halfs), 32B-mult
  __shared__ _Float16 Ksm[64][KS];             // Ksm[key][dk]   (== B for QK^T)
  __shared__ _Float16 Vsm[64][KS];             // Vsm[dv][key]   (== B for P*V)
  __shared__ _Float16 Psm[8][16][KS];          // per-wave P (row, key)
  const int tid = threadIdx.x, lane = tid & 31, wave = tid >> 5;
  const int hi = lane >> 4, nl = lane & 15;
  const int b = blockIdx.y >> 4, h = blockIdx.y & 15;
  const int qrow = blockIdx.x * 128 + wave * 16;
  const size_t headoff = (size_t)b * S * 1024 + (size_t)h * 64;

  v16h qf[2];
#pragma unroll
  for (int t = 0; t < 2; ++t)
    qf[t] = load_frag_a(qh + headoff + (size_t)qrow * 1024 + t * 32, 1024, lane);

  v8f o[4];
#pragma unroll
  for (int nt = 0; nt < 4; ++nt) o[nt] = (v8f)0.0f;
  float mrow[8], lrow[8];
#pragma unroll
  for (int r = 0; r < 8; ++r) { mrow[r] = -1e30f; lrow[r] = 0.0f; }

  const int kr = tid >> 2, c0 = (tid & 3) * 16;   // K stage: 64 keys x 64 dk
  const int kp = tid >> 3, d0 = (tid & 7) * 8;    // V stage: 32 key-pairs x 8 dv

  for (int kb = 0; kb < S; kb += 64) {
    __syncthreads();
    {
      // --- K: straight f16 copy, async global->LDS (b128 = 16B per lane) ---
      const _Float16* ksrc = kh + headoff + (size_t)(kb + kr) * 1024 + c0;
      unsigned kdst = lds_off(&Ksm[kr][c0]);
      unsigned long long ga = (unsigned long long)(uintptr_t)ksrc;
      asm volatile("global_load_async_to_lds_b128 %0, %1, off"
                   :: "v"(kdst), "v"(ga) : "memory");
      asm volatile("global_load_async_to_lds_b128 %0, %1, off offset:16"
                   :: "v"(kdst), "v"(ga) : "memory");
      // --- V: transpose stage, pack 2 keys per b32 store ---
      const _Float16* v0 = vh + headoff + (size_t)(kb + 2 * kp) * 1024 + d0;
      const _Float16* v1 = v0 + 1024;
      v8h a = *(const v8h*)(v0);
      v8h c = *(const v8h*)(v1);
#pragma unroll
      for (int j = 0; j < 8; ++j)
        *(unsigned*)&Vsm[d0 + j][2 * kp] = pack2h(a[j], c[j]);
    }
    asm volatile("s_wait_asynccnt 0x0" ::: "memory");
    __syncthreads();

    // S = Q K^T / sqrt(64)
    v8f s[4];
#pragma unroll
    for (int kt = 0; kt < 4; ++kt) {
      v8f sc = (v8f)0.0f;
      sc = wmma_f16(qf[0], load_frag_b(&Ksm[kt * 16][0], KS, lane), sc);
      sc = wmma_f16(qf[1], load_frag_b(&Ksm[kt * 16][0] + 32, KS, lane), sc);
      s[kt] = sc * 0.125f;
    }

    // online softmax: rows live at element r (m = r + 8*hi), cols across lanes
    float mnew[8], rsum[8], alpha[8];
#pragma unroll
    for (int r = 0; r < 8; ++r) {
      float v = fmaxf(fmaxf(s[0][r], s[1][r]), fmaxf(s[2][r], s[3][r]));
#pragma unroll
      for (int off = 8; off >= 1; off >>= 1) v = fmaxf(v, __shfl_xor(v, off, 32));
      mnew[r] = fmaxf(mrow[r], v);
      rsum[r] = 0.0f;
    }
#pragma unroll
    for (int kt = 0; kt < 4; ++kt)
#pragma unroll
      for (int r = 0; r < 8; ++r) {
        float p = __expf(s[kt][r] - mnew[r]);
        rsum[r] += p;
        Psm[wave][r + hi * 8][kt * 16 + nl] = (_Float16)p;
      }
#pragma unroll
    for (int r = 0; r < 8; ++r) {
#pragma unroll
      for (int off = 8; off >= 1; off >>= 1) rsum[r] += __shfl_xor(rsum[r], off, 32);
      alpha[r] = __expf(mrow[r] - mnew[r]);
      lrow[r] = lrow[r] * alpha[r] + rsum[r];
      mrow[r] = mnew[r];
    }
#pragma unroll
    for (int nt = 0; nt < 4; ++nt)
#pragma unroll
      for (int r = 0; r < 8; ++r) o[nt][r] *= alpha[r];

    __syncthreads();   // Psm visible (D->A re-layout through LDS)

    v16h pf[2];
#pragma unroll
    for (int t = 0; t < 2; ++t) pf[t] = load_frag_a(&Psm[wave][0][0] + t * 32, KS, lane);
#pragma unroll
    for (int nt = 0; nt < 4; ++nt) {
      o[nt] = wmma_f16(pf[0], load_frag_b(&Vsm[nt * 16][0], KS, lane), o[nt]);
      o[nt] = wmma_f16(pf[1], load_frag_b(&Vsm[nt * 16][0] + 32, KS, lane), o[nt]);
    }
  }

#pragma unroll
  for (int nt = 0; nt < 4; ++nt)
#pragma unroll
    for (int r = 0; r < 8; ++r) {
      float val = o[nt][r] / lrow[r];
      outh[headoff + (size_t)(qrow + r + hi * 8) * 1024 + nt * 16 + nl] = (_Float16)val;
    }
}

// ----------------------------------------------------------- LayerNorm ----
__global__ __launch_bounds__(256) void ln_kernel(
    const float* __restrict__ x, const float* __restrict__ gamma,
    const float* __restrict__ beta, float* __restrict__ out, int N) {
  const int row = blockIdx.x, tid = threadIdx.x, lane = tid & 31, wave = tid >> 5;
  const float* xr = x + (size_t)row * N;
  float s = 0.0f, s2 = 0.0f;
  for (int i = tid; i < N; i += 256) { float v = xr[i]; s += v; s2 += v * v; }
#pragma unroll
  for (int off = 16; off >= 1; off >>= 1) {
    s += __shfl_xor(s, off, 32);
    s2 += __shfl_xor(s2, off, 32);
  }
  __shared__ float red[2][8];
  if (lane == 0) { red[0][wave] = s; red[1][wave] = s2; }
  __syncthreads();
  float ts = 0.0f, ts2 = 0.0f;
#pragma unroll
  for (int w = 0; w < 8; ++w) { ts += red[0][w]; ts2 += red[1][w]; }
  const float mu = ts / (float)N;
  const float var = ts2 / (float)N - mu * mu;
  const float rstd = rsqrtf(var + 1e-6f);
  for (int i = tid; i < N; i += 256)
    out[(size_t)row * N + i] = (xr[i] - mu) * rstd * gamma[i] + beta[i];
}

// ------------------------------------------------------------- launch -----
extern "C" void kernel_launch(void* const* d_in, const int* in_sizes, int n_in,
                              void* d_out, int out_size, void* d_ws, size_t ws_size,
                              hipStream_t stream) {
  (void)in_sizes; (void)n_in; (void)out_size; (void)ws_size;
  const float* q     = (const float*)d_in[0];
  const float* k     = (const float*)d_in[1];
  const float* v     = (const float*)d_in[2];
  const float* Wq    = (const float*)d_in[3];
  const float* Wk    = (const float*)d_in[4];
  const float* Wv    = (const float*)d_in[5];
  const float* Wfc   = (const float*)d_in[6];
  const float* bfc   = (const float*)d_in[7];
  const float* gamma = (const float*)d_in[8];
  const float* beta  = (const float*)d_in[9];
  float* out = (float*)d_out;

  const int S = 2048, D = 1024, M = 2 * S;   // B=2

  // workspace: qh,kh,vh,attn (f16) then pre-LN x (f32)  -> ~48 MB
  _Float16* qh = (_Float16*)d_ws;
  _Float16* kh = qh + (size_t)M * D;
  _Float16* vh = kh + (size_t)M * D;
  _Float16* at = vh + (size_t)M * D;
  float*  xres = (float*)(at + (size_t)M * D);

  dim3 blk(256);
  dim3 gg(D / 128, M / 128);                 // 8 x 32
  gemm_kernel<false, false><<<gg, blk, 0, stream>>>(q, Wq, qh, nullptr, nullptr, nullptr, M, D, D);
  gemm_kernel<false, false><<<gg, blk, 0, stream>>>(k, Wk, kh, nullptr, nullptr, nullptr, M, D, D);
  gemm_kernel<false, false><<<gg, blk, 0, stream>>>(v, Wv, vh, nullptr, nullptr, nullptr, M, D, D);

  attn_kernel<<<dim3(S / 128, 32), blk, 0, stream>>>(qh, kh, vh, at, S);

  gemm_kernel<true, true><<<gg, blk, 0, stream>>>(at, Wfc, nullptr, xres, bfc, q, M, D, D);

  ln_kernel<<<dim3(M), blk, 0, stream>>>(xres, gamma, beta, out, D);
}
